// MTRNNCell_46299747451485
// MI455X (gfx1250) — matrix operand order repaired and checked by reading, not compiled, not run on hardware
//
#include <hip/hip_runtime.h>
#include <cstdint>

typedef __attribute__((ext_vector_type(16))) _Float16 v16h;
typedef __attribute__((ext_vector_type(8)))  float    v8f;
typedef __attribute__((ext_vector_type(4)))  int      v4i;

#define B_ROWS 16384
#define K_CAT  1280   // 512 (fast) + 256 (slow) + 512 (io)
#define N_FAST 512
#define N_OUT  64
#define K_IO   512

union Frag16 { v16h v; uint4 q[2]; };

__device__ __forceinline__ float sigf(float x) { return 1.0f / (1.0f + expf(-x)); }

// ---- CDNA5 async global->LDS path (guarded; fallback = batched sync copy) ----
#if defined(__has_builtin)
#if __has_builtin(__builtin_amdgcn_global_load_async_to_lds_b128)
#define HAVE_ASYNC_LDS 1
#endif
#endif

#ifdef HAVE_ASYNC_LDS
// builtin expects (int4 global*, int4 lds*, imm offset, imm cpol)
#define ASYNC_COPY_B128(gp, lp)                                                     \
  __builtin_amdgcn_global_load_async_to_lds_b128(                                   \
      (__attribute__((address_space(1))) v4i*)(gp),                                 \
      (__attribute__((address_space(3))) v4i*)(lp), 0, 0)
#if __has_builtin(__builtin_amdgcn_s_wait_asynccnt)
#define WAIT_ASYNC0() __builtin_amdgcn_s_wait_asynccnt(0)
#else
#define WAIT_ASYNC0() asm volatile("s_wait_asynccnt 0x0" ::: "memory")
#endif
#else
#define WAIT_ASYNC0() ((void)0)
#endif

// ---------------- pack / elementwise kernels ----------------

// Xcat[b, 0:512]=fast, [512:768]=slow, [768:1280]=io  (f32 -> f16)
__global__ __launch_bounds__(256) void k_pack_x(const float* __restrict__ io,
                                                const float* __restrict__ fast,
                                                const float* __restrict__ slow,
                                                _Float16* __restrict__ X) {
  size_t idx = (size_t)blockIdx.x * 256 + threadIdx.x;
  if (idx >= (size_t)B_ROWS * K_CAT) return;
  int row = (int)(idx / K_CAT), col = (int)(idx % K_CAT);
  float v;
  if (col < 512)       v = fast[(size_t)row * 512 + col];
  else if (col < 768)  v = slow[(size_t)row * 256 + (col - 512)];
  else                 v = io  [(size_t)row * 512 + (col - 768)];
  X[idx] = (_Float16)v;
}

// Wcat[n, 0:512]=W_f2f[n], [512:768]=W_s2f[n], [768:1280]=W_in2f[n]
__global__ __launch_bounds__(256) void k_pack_wcat(const float* __restrict__ Wf2f,
                                                   const float* __restrict__ Ws2f,
                                                   const float* __restrict__ Win2f,
                                                   _Float16* __restrict__ W) {
  size_t idx = (size_t)blockIdx.x * 256 + threadIdx.x;
  if (idx >= (size_t)N_FAST * K_CAT) return;
  int row = (int)(idx / K_CAT), col = (int)(idx % K_CAT);
  float v;
  if (col < 512)       v = Wf2f [(size_t)row * 512 + col];
  else if (col < 768)  v = Ws2f [(size_t)row * 256 + (col - 512)];
  else                 v = Win2f[(size_t)row * 512 + (col - 768)];
  W[idx] = (_Float16)v;
}

__global__ __launch_bounds__(256) void k_pack_f16(const float* __restrict__ src,
                                                  _Float16* __restrict__ dst, size_t n) {
  size_t idx = (size_t)blockIdx.x * 256 + threadIdx.x;
  if (idx < n) dst[idx] = (_Float16)src[idx];
}

__global__ __launch_bounds__(256) void k_bsum(const float* __restrict__ a,
                                              const float* __restrict__ b,
                                              const float* __restrict__ c,
                                              float* __restrict__ o) {
  int i = blockIdx.x * 256 + threadIdx.x;
  if (i < N_FAST) o[i] = a[i] + b[i] + c[i];
}

__global__ __launch_bounds__(256) void k_sigmoid(const float* __restrict__ in,
                                                 float* __restrict__ out, size_t n) {
  size_t idx = (size_t)blockIdx.x * 256 + threadIdx.x;
  if (idx < n) out[idx] = sigf(in[idx]);
}

// ---------------- WMMA GEMM, double-buffered async LDS staging ----------------
// out[m,n] = f( sum_k A[m,k]*W[n,k] + bias[n] )
// MODE 0: sigmoid(0.8*fast[m,n] + (acc + bias[n])*0.2)   (new_fast)
// MODE 1: tanh(acc + bias[n])                            (output head)
// 256 threads = 8 waves in a 4(M) x 2(N) grid; per-wave MT x NT 16x16 tiles.
template<int BM, int BN, int MT, int NT, int MODE>
__global__ __launch_bounds__(256) void k_gemm(const _Float16* __restrict__ A, int lda,
                                              const _Float16* __restrict__ W, int ldw, int K,
                                              const float* __restrict__ bias,
                                              const float* __restrict__ fast,
                                              float* __restrict__ out, int ldo) {
  __shared__ _Float16 As[2][BM * 32];
  __shared__ _Float16 Bs[2][BN * 32];
  constexpr int QA = (BM * 4) / 256;  // 16B chunks per thread for A tile
  constexpr int QB = (BN * 4) / 256;  // 16B chunks per thread for B tile

  const int tid  = threadIdx.x;
  const int gm   = blockIdx.x * BM;
  const int gn   = blockIdx.y * BN;
  const int wave = tid >> 5;
  const int lane = tid & 31;
  const int half = lane >> 4;
  const int lrow = lane & 15;
  const int wm   = (wave >> 1) * (MT * 16);
  const int wn   = (wave & 1)  * (NT * 16);

  // issue staging for K-tile at k0 into buffer `buf`
  auto stage = [&](int buf, int k0) {
#ifdef HAVE_ASYNC_LDS
#pragma unroll
    for (int j = 0; j < QA; ++j) {
      int idx = tid + j * 256, r = idx >> 2, q = idx & 3;
      ASYNC_COPY_B128(A + (size_t)(gm + r) * lda + k0 + q * 8,
                      &As[buf][r * 32 + q * 8]);
    }
#pragma unroll
    for (int j = 0; j < QB; ++j) {
      int idx = tid + j * 256, r = idx >> 2, q = idx & 3;
      ASYNC_COPY_B128(W + (size_t)(gn + r) * ldw + k0 + q * 8,
                      &Bs[buf][r * 32 + q * 8]);
    }
#else
    uint4 ta[QA], tb[QB];
#pragma unroll
    for (int j = 0; j < QA; ++j) {
      int idx = tid + j * 256, r = idx >> 2, q = idx & 3;
      ta[j] = *(const uint4*)(A + (size_t)(gm + r) * lda + k0 + q * 8);
    }
#pragma unroll
    for (int j = 0; j < QB; ++j) {
      int idx = tid + j * 256, r = idx >> 2, q = idx & 3;
      tb[j] = *(const uint4*)(W + (size_t)(gn + r) * ldw + k0 + q * 8);
    }
#pragma unroll
    for (int j = 0; j < QA; ++j) {
      int idx = tid + j * 256, r = idx >> 2, q = idx & 3;
      *(uint4*)(&As[buf][r * 32 + q * 8]) = ta[j];
    }
#pragma unroll
    for (int j = 0; j < QB; ++j) {
      int idx = tid + j * 256, r = idx >> 2, q = idx & 3;
      *(uint4*)(&Bs[buf][r * 32 + q * 8]) = tb[j];
    }
#endif
  };

  v8f acc[MT][NT];
#pragma unroll
  for (int i = 0; i < MT; ++i)
#pragma unroll
    for (int j = 0; j < NT; ++j)
#pragma unroll
      for (int r = 0; r < 8; ++r) acc[i][j][r] = 0.0f;

  const int niter = K / 32;
  stage(0, 0);  // prologue fill

  for (int i = 0; i < niter; ++i) {
    const int cur = i & 1;
    WAIT_ASYNC0();          // own async fills (buffer cur) landed in LDS
    __syncthreads();        // everyone's fills landed; prior reads of cur^1 done
    if (i + 1 < niter) stage(cur ^ 1, (i + 1) * 32);  // overlap fill with compute

    // A fragments: lane lrow holds row (wm+tm*16+lrow); K chunks per ISA layout
    Frag16 af[MT];
#pragma unroll
    for (int tm = 0; tm < MT; ++tm) {
      int row = wm + tm * 16 + lrow;
      af[tm].q[0] = *(const uint4*)(&As[cur][row * 32 + half * 8]);
      af[tm].q[1] = *(const uint4*)(&As[cur][row * 32 + 16 + half * 8]);
    }
    // B fragments: lane holds column (wn+tn*16+lrow); K chunks per ISA layout
    Frag16 bf[NT];
#pragma unroll
    for (int tn = 0; tn < NT; ++tn) {
      int col = wn + tn * 16 + lrow;
      bf[tn].q[0] = *(const uint4*)(&Bs[cur][col * 32 + half * 8]);
      bf[tn].q[1] = *(const uint4*)(&Bs[cur][col * 32 + 16 + half * 8]);
    }
#pragma unroll
    for (int tm = 0; tm < MT; ++tm)
#pragma unroll
      for (int tn = 0; tn < NT; ++tn)
        acc[tm][tn] = __builtin_amdgcn_wmma_f32_16x16x32_f16(
            false, af[tm].v, false, bf[tn].v, (short)0, acc[tm][tn], false, false);
  }

  // epilogue: C/D layout -> element r is (M = r + 8*half, N = lrow)
#pragma unroll
  for (int tm = 0; tm < MT; ++tm)
#pragma unroll
    for (int tn = 0; tn < NT; ++tn) {
      int baseR = gm + wm + tm * 16 + half * 8;
      int c     = gn + wn + tn * 16 + lrow;
      float bn  = bias[c];
#pragma unroll
      for (int r = 0; r < 8; ++r) {
        int   row = baseR + r;
        float v   = acc[tm][tn][r];
        if (MODE == 0) {
          float f = fast[(size_t)row * N_FAST + c];
          v = sigf(0.8f * f + (v + bn) * 0.2f);
        } else {
          v = tanhf(v + bn);
        }
        out[(size_t)row * ldo + c] = v;
      }
    }
}

// ---------------- launcher ----------------

extern "C" void kernel_launch(void* const* d_in, const int* in_sizes, int n_in,
                              void* d_out, int out_size, void* d_ws, size_t ws_size,
                              hipStream_t stream) {
  // input order (setup_inputs):
  // 0 input, 1 io_h, 2 fast_h, 3 slow_h,
  // 4 W_in2io, 5 b_in2io, 6 W_in2f, 7 b_in2f, 8 W_f2f, 9 b_f2f,
  // 10 W_f2s, 11 b_f2s, 12 W_s2s, 13 b_s2s, 14 W_s2f, 15 b_s2f,
  // 16 W_f2io, 17 b_f2io, 18 W_io2io, 19 b_io2io, 20 W_io2out, 21 b_io2out
  const float* io_h    = (const float*)d_in[1];
  const float* fast_h  = (const float*)d_in[2];
  const float* slow_h  = (const float*)d_in[3];
  const float* W_in2f  = (const float*)d_in[6];
  const float* b_in2f  = (const float*)d_in[7];
  const float* W_f2f   = (const float*)d_in[8];
  const float* b_f2f   = (const float*)d_in[9];
  const float* W_s2f   = (const float*)d_in[14];
  const float* b_s2f   = (const float*)d_in[15];
  const float* W_io2o  = (const float*)d_in[20];
  const float* b_io2o  = (const float*)d_in[21];

  float* out = (float*)d_out;
  float* out0 = out;                                   // (B,64)  tanh output
  float* out1 = out0 + (size_t)B_ROWS * N_OUT;         // (B,512) sigmoid(io_h)
  float* out2 = out1 + (size_t)B_ROWS * 512;           // (B,512) sigmoid(new_fast)
  float* out3 = out2 + (size_t)B_ROWS * 512;           // (B,256) sigmoid(slow_h)

  // workspace layout
  char* ws = (char*)d_ws;
  _Float16* Xcat = (_Float16*)ws;                                        // B x 1280
  _Float16* Wcat = (_Float16*)(ws + (size_t)B_ROWS * K_CAT * 2);         // 512 x 1280
  _Float16* Wout = (_Float16*)((char*)Wcat + (size_t)N_FAST * K_CAT * 2);// 64 x 512
  float*    bsum = (float*)((char*)Wout + (size_t)N_OUT * K_IO * 2);     // 512

  // conversions / packing
  k_pack_x<<<(B_ROWS * (size_t)K_CAT) / 256, 256, 0, stream>>>(io_h, fast_h, slow_h, Xcat);
  k_pack_wcat<<<(N_FAST * K_CAT) / 256, 256, 0, stream>>>(W_f2f, W_s2f, W_in2f, Wcat);
  k_pack_f16<<<(N_OUT * K_IO) / 256, 256, 0, stream>>>(W_io2o, Wout, (size_t)N_OUT * K_IO);
  k_bsum<<<2, 256, 0, stream>>>(b_f2f, b_s2f, b_in2f, bsum);

  // elementwise pass-through outputs
  k_sigmoid<<<((size_t)B_ROWS * 512) / 256, 256, 0, stream>>>(io_h, out1, (size_t)B_ROWS * 512);
  k_sigmoid<<<((size_t)B_ROWS * 256) / 256, 256, 0, stream>>>(slow_h, out3, (size_t)B_ROWS * 256);

  // new_fast: (B x 512) = Xcat(B x 1280) @ Wcat^T, fused leak + sigmoid
  dim3 g0(B_ROWS / 128, N_FAST / 128);
  k_gemm<128, 128, 2, 4, 0><<<g0, 256, 0, stream>>>(
      Xcat, K_CAT, Wcat, K_CAT, K_CAT, bsum, fast_h, out2, N_FAST);

  // output head: (B x 64) = io_h16(B x 512) @ Wout^T, fused tanh
  // io_h in f16 lives at Xcat columns 768..1279 (lda = 1280)
  dim3 g1(B_ROWS / 128, 1);
  k_gemm<128, 64, 2, 2, 1><<<g1, 256, 0, stream>>>(
      Xcat + 768, K_CAT, Wout, K_IO, K_IO, b_io2o, nullptr, out0, N_OUT);
}
